// CrossAttention_65103114273038
// MI455X (gfx1250) — compile-verified, matrix operand order checked by reading, not run on hardware
//
#include <hip/hip_runtime.h>

typedef __attribute__((ext_vector_type(16))) _Float16 v16h;
typedef __attribute__((ext_vector_type(8)))  _Float16 v8h;
typedef __attribute__((ext_vector_type(4)))  _Float16 v4h;
typedef __attribute__((ext_vector_type(8)))  float    v8f;
typedef __attribute__((ext_vector_type(4)))  float    v4f;
typedef __attribute__((ext_vector_type(4)))  unsigned int u32x4;
typedef __attribute__((ext_vector_type(8)))  int      i32x8;
typedef __attribute__((ext_vector_type(4)))  int      i32x4;

// Address-space-qualified vector pointers for the async LDS builtins
// (signature probed: (int4 AS1*, int4 AS3*, imm offset, imm cpol)).
typedef __attribute__((address_space(1))) i32x4 gvec4_t;
typedef __attribute__((address_space(3))) i32x4 lvec4_t;

#define WMMA_F16(a, b, c) \
  __builtin_amdgcn_wmma_f32_16x16x32_f16(false, (a), false, (b), (short)0, (c), false, false)

// Feature detection (evaluated independently for host/device passes).
#if defined(__gfx1250__) && __has_builtin(__builtin_amdgcn_global_load_async_to_lds_b128)
#define HAVE_ASYNC_LDS 1
#else
#define HAVE_ASYNC_LDS 0
#endif
#if defined(__gfx1250__) && __has_builtin(__builtin_amdgcn_tensor_load_to_lds)
#define HAVE_TDM 1
#else
#define HAVE_TDM 0
#endif

#if HAVE_ASYNC_LDS
// AS1: identical bit pattern to the flat address (int->ptr cast only).
// AS3: flat LDS addresses carry the LDS byte offset in the low 32 bits
// (ISA 10.2).
__device__ __forceinline__ void async_cp16(const void* gsrc, void* ldst) {
  __builtin_amdgcn_global_load_async_to_lds_b128(
      (gvec4_t*)(unsigned long long)(uintptr_t)gsrc,
      (lvec4_t*)(unsigned)(uintptr_t)ldst, 0, 0);
}
__device__ __forceinline__ void async_wait0() {
#if __has_builtin(__builtin_amdgcn_s_wait_asynccnt)
  __builtin_amdgcn_s_wait_asynccnt(0);
#else
  asm volatile("s_wait_asynccnt 0" ::: "memory");
#endif
}
#endif

// ---------------------------------------------------------------------------
// 16-bit WMMA fragment load from a row-major LDS tile.
// Serves A-fragments (x = row M) and B-fragments when the tile is stored
// transposed (x = col N).  Per CDNA5 ISA 7.12.2 (16-bit A 16x32):
//   lane L: x = L&15, kbase = (L>>4)*8
//   half i (0..15): k = ((i&8)?16:0) + kbase + (i&7)
// -> two contiguous 16-byte LDS reads per fragment.
// ---------------------------------------------------------------------------
__device__ __forceinline__ v16h frag_ld(const _Float16* p, int ld) {
  const int lane = threadIdx.x & 31;
  const int x    = lane & 15;
  const int kb   = (lane >> 4) << 3;
  const _Float16* r = p + x * ld + kb;
  v8h lo = *(const v8h*)(r);
  v8h hi = *(const v8h*)(r + 16);
  v16h out;
#pragma unroll
  for (int i = 0; i < 8; ++i) { out[i] = lo[i]; out[i + 8] = hi[i]; }
  return out;
}

// ---------------------------------------------------------------------------
// GEMM: C[M,N] = A[M,K](f32) * W[K,N](f32, row-major) + bias (+ residual)
// Tiles: 128(M) x 64(N) per 256-thread block (8 waves, 4x2 wave grid,
// 32x32 per wave), BK = 32 (one WMMA K-step).  fp32 -> f16 conversion
// happens during LDS staging; accumulation in f32.
// ---------------------------------------------------------------------------
template <bool OUT_F16>
__global__ __launch_bounds__(256) void gemm_bias_kernel(
    const float* __restrict__ A, int lda,
    const float* __restrict__ W, int ldw,
    const float* __restrict__ bias,
    const float* __restrict__ residual,  // nullable, leading dim = ldc
    void* __restrict__ Cout, int ldc, int K) {
  __shared__ __align__(16) _Float16 As[128 * 32];   // [m][k]
  __shared__ __align__(16) _Float16 WsT[64 * 32];   // [n][k] (transposed)

  const int t    = threadIdx.x;
  const int row0 = blockIdx.y * 128;
  const int n0   = blockIdx.x * 64;
  const int wave = t >> 5;
  const int wm   = wave >> 1;  // 0..3  (M sub-tile)
  const int wn   = wave & 1;   // 0..1  (N sub-tile)

  v8f acc[2][2] = {};

  for (int kk = 0; kk < K; kk += 32) {
    __syncthreads();
    // Stage A tile (128x32 f32 -> f16)
#pragma unroll
    for (int c = 0; c < 4; ++c) {
      int idx = (c * 256 + t) * 4;
      int r = idx >> 5, col = idx & 31;
      v4f v = *(const v4f*)(A + (size_t)(row0 + r) * lda + kk + col);
      v4h hv;
#pragma unroll
      for (int j = 0; j < 4; ++j) hv[j] = (_Float16)v[j];
      *(v4h*)(As + r * 32 + col) = hv;
    }
    // Stage W tile transposed (32x64 f32 -> f16, stored [n][k])
#pragma unroll
    for (int c = 0; c < 2; ++c) {
      int idx = (c * 256 + t) * 4;
      int kq = idx >> 6, n = idx & 63;
      v4f v = *(const v4f*)(W + (size_t)(kk + kq) * ldw + n0 + n);
#pragma unroll
      for (int j = 0; j < 4; ++j) WsT[(n + j) * 32 + kq] = (_Float16)v[j];
    }
    if (kk + 32 < K)  // speculative prefetch of next A tile -> global_prefetch_b8
      __builtin_prefetch(A + (size_t)(row0 + (t >> 1)) * lda + kk + 32, 0, 1);
    __syncthreads();

    v16h a0 = frag_ld(As + (wm * 32 + 0) * 32, 32);
    v16h a1 = frag_ld(As + (wm * 32 + 16) * 32, 32);
    v16h b0 = frag_ld(WsT + (wn * 32 + 0) * 32, 32);
    v16h b1 = frag_ld(WsT + (wn * 32 + 16) * 32, 32);
    acc[0][0] = WMMA_F16(a0, b0, acc[0][0]);
    acc[0][1] = WMMA_F16(a0, b1, acc[0][1]);
    acc[1][0] = WMMA_F16(a1, b0, acc[1][0]);
    acc[1][1] = WMMA_F16(a1, b1, acc[1][1]);
  }

  // Epilogue: C layout -> lane holds col n = lane&15, rows e + 8*(lane>>4)
  const int lane = t & 31;
  const int nl = lane & 15, hg = lane >> 4;
#pragma unroll
  for (int mi = 0; mi < 2; ++mi)
#pragma unroll
    for (int nj = 0; nj < 2; ++nj)
#pragma unroll
      for (int e = 0; e < 8; ++e) {
        size_t row = (size_t)row0 + wm * 32 + mi * 16 + e + 8 * hg;
        int col = n0 + wn * 32 + nj * 16 + nl;
        float v = acc[mi][nj][e] + bias[col];
        if (residual) v += residual[row * ldc + col];
        if constexpr (OUT_F16)
          ((_Float16*)Cout)[row * ldc + col] = (_Float16)v;
        else
          ((float*)Cout)[row * ldc + col] = v;
      }
}

// ---------------------------------------------------------------------------
// Flash attention.  Grid: (Sq/128, H, B), 128 threads = 4 waves.
// Wave w owns 32 q-rows.  K-tiles of 64 keys.  Online softmax, scale = d^-0.5.
// Q: f16 [B*Sq, 512] (head h at cols h*64..), KV: f16 [B*Skv, 1024]
// (K cols 0..511, V cols 512..1023).  Output O: f32 [B*Sq, 512].
// K tile is fetched by the Tensor Data Mover (descriptor-driven DMA) when the
// builtin is available; Q tile uses async global->LDS copies.
// ---------------------------------------------------------------------------
__global__ __launch_bounds__(128) void attention_kernel(
    const _Float16* __restrict__ Q, const _Float16* __restrict__ KV,
    float* __restrict__ O) {
  __shared__ __align__(16) _Float16 Qs[128 * 64];       // [qrow][d]
  __shared__ __align__(16) _Float16 Ks[64 * 64];        // [key][d]
  __shared__ __align__(16) _Float16 VsT[64 * 64];       // [d][key]
  __shared__ __align__(16) _Float16 Ps[4 * 32 * 64];    // per-wave P [row][key]

  const int t = threadIdx.x;
  const int wave = t >> 5, lane = t & 31;
  const int nl = lane & 15, hg = lane >> 4;
  const int b = blockIdx.z, h = blockIdx.y;
  const size_t qrb = (size_t)b * 2048 + blockIdx.x * 128;

  // Load the 128x64 Q tile once
#pragma unroll
  for (int c = 0; c < 8; ++c) {
    int idx = (c * 128 + t) * 8;
    int r = idx >> 6, col = idx & 63;
#if HAVE_ASYNC_LDS
    async_cp16(Q + (qrb + r) * 512 + h * 64 + col, Qs + idx);
#else
    *(v8h*)(Qs + idx) = *(const v8h*)(Q + (qrb + r) * 512 + h * 64 + col);
#endif
  }
#if HAVE_ASYNC_LDS
  async_wait0();
#endif
  __syncthreads();

  v16h aq[2][2];
#pragma unroll
  for (int mi = 0; mi < 2; ++mi)
#pragma unroll
    for (int kd = 0; kd < 2; ++kd)
      aq[mi][kd] = frag_ld(Qs + (wave * 32 + mi * 16) * 64 + kd * 32, 64);

  v8f o[2][4] = {};
  float mrow[2][8], lrow[2][8];
#pragma unroll
  for (int mi = 0; mi < 2; ++mi)
#pragma unroll
    for (int e = 0; e < 8; ++e) { mrow[mi][e] = -3.0e38f; lrow[mi][e] = 0.f; }

  for (int tk = 0; tk < 2048; tk += 64) {
    __syncthreads();
    const _Float16* ktile = KV + ((size_t)b * 2048 + tk) * 1024 + h * 64;

#if HAVE_TDM
    // --- K tile via Tensor Data Mover: 64x64 f16 tile, row stride 1024 elems.
    if (wave == 0) {
      unsigned long long ga = (unsigned long long)(uintptr_t)ktile;
      u32x4 g0;
      g0[0] = 1u;                                   // count=1, user mode
      g0[1] = (unsigned)(uintptr_t)Ks;              // lds_addr (bytes)
      g0[2] = (unsigned)(ga & 0xFFFFFFFFull);       // global_addr[31:0]
      g0[3] = (unsigned)((ga >> 32) & 0x1FFFFFFull) // global_addr[56:32]
              | (2u << 30);                         // type = 2 ("image")
      i32x8 g1;
      g1[0] = (1 << 16);          // data_size = 1 (2 bytes), no pad/iterate
      g1[1] = (64 << 16);         // tensor_dim0[15:0] = 64 (in bits 63:48)
      g1[2] = (64 << 16);         // tensor_dim0[31:16]=0 | tensor_dim1[15:0]=64
      g1[3] = (64 << 16);         // tensor_dim1[31:16]=0 | tile_dim0=64
      g1[4] = 64;                 // tile_dim1=64 | tile_dim2=0
      g1[5] = 1024;               // tensor_dim0_stride[31:0] = 1024 elems
      g1[6] = 0;                  // stride0[47:32]=0 | stride1[15:0]=0
      g1[7] = 0;
      i32x4 z4 = {0, 0, 0, 0};
      i32x8 z8 = {0, 0, 0, 0, 0, 0, 0, 0};
      __builtin_amdgcn_tensor_load_to_lds(g0, g1, z4, z4, z8, 0);
    }
#elif HAVE_ASYNC_LDS
#pragma unroll
    for (int c = 0; c < 4; ++c) {
      int idx = (c * 128 + t) * 8;
      int key = idx >> 6, col = idx & 63;
      async_cp16(ktile + (size_t)key * 1024 + col, Ks + idx);
    }
#else
#pragma unroll
    for (int c = 0; c < 4; ++c) {
      int idx = (c * 128 + t) * 8;
      int key = idx >> 6, col = idx & 63;
      *(v8h*)(Ks + idx) = *(const v8h*)(ktile + (size_t)key * 1024 + col);
    }
#endif

    // --- V tile, transposed into [d][key] (manual: needs transpose)
#pragma unroll
    for (int c = 0; c < 4; ++c) {
      int idx = (c * 128 + t) * 8;
      int key = idx >> 6, col = idx & 63;
      v8h vv = *(const v8h*)(ktile + (size_t)key * 1024 + 512 + col);
#pragma unroll
      for (int j = 0; j < 8; ++j) VsT[(col + j) * 64 + key] = vv[j];
    }

#if HAVE_TDM
    if (wave == 0) __builtin_amdgcn_s_wait_tensorcnt(0);
#elif HAVE_ASYNC_LDS
    async_wait0();
#endif
    __syncthreads();

    _Float16* Pw = Ps + wave * (32 * 64);
#pragma unroll
    for (int mi = 0; mi < 2; ++mi) {
      // S = (Q K^T) * scale  — B-fragment from Ks is contiguous: elem(k,n)=Ks[n][k]
      v8f s[4];
#pragma unroll
      for (int nj = 0; nj < 4; ++nj) {
        v8f z = {};
        z = WMMA_F16(aq[mi][0], frag_ld(Ks + nj * 16 * 64 + 0, 64), z);
        z = WMMA_F16(aq[mi][1], frag_ld(Ks + nj * 16 * 64 + 32, 64), z);
#pragma unroll
        for (int e = 0; e < 8; ++e) s[nj][e] = z[e] * 0.125f;
      }
      // Row max across 4 n-frags + 16-lane group
      float tmax[8];
#pragma unroll
      for (int e = 0; e < 8; ++e)
        tmax[e] = fmaxf(fmaxf(s[0][e], s[1][e]), fmaxf(s[2][e], s[3][e]));
#pragma unroll
      for (int off = 1; off < 16; off <<= 1)
#pragma unroll
        for (int e = 0; e < 8; ++e)
          tmax[e] = fmaxf(tmax[e], __shfl_xor(tmax[e], off, 32));
      float alpha[8];
#pragma unroll
      for (int e = 0; e < 8; ++e) {
        float mo = mrow[mi][e];
        float mn = fmaxf(mo, tmax[e]);
        mrow[mi][e] = mn;
        alpha[e] = __expf(mo - mn);
      }
      // P = exp(S - m); write C-layout -> LDS (A-layout re-read below)
      float tsum[8] = {0, 0, 0, 0, 0, 0, 0, 0};
#pragma unroll
      for (int nj = 0; nj < 4; ++nj)
#pragma unroll
        for (int e = 0; e < 8; ++e) {
          float p = __expf(s[nj][e] - mrow[mi][e]);
          tsum[e] += p;
          Pw[(mi * 16 + e + 8 * hg) * 64 + nj * 16 + nl] = (_Float16)p;
        }
#pragma unroll
      for (int off = 1; off < 16; off <<= 1)
#pragma unroll
        for (int e = 0; e < 8; ++e) tsum[e] += __shfl_xor(tsum[e], off, 32);
#pragma unroll
      for (int e = 0; e < 8; ++e) lrow[mi][e] = lrow[mi][e] * alpha[e] + tsum[e];
#pragma unroll
      for (int nj = 0; nj < 4; ++nj)
#pragma unroll
        for (int e = 0; e < 8; ++e) o[mi][nj][e] *= alpha[e];
    }
    // LDS is in-order per wave; fence the compiler + drain DScnt explicitly
    asm volatile("s_wait_dscnt 0" ::: "memory");
    // O += P * V
#pragma unroll
    for (int mi = 0; mi < 2; ++mi) {
      v16h p0 = frag_ld(Ps + wave * 2048 + mi * 16 * 64 + 0, 64);
      v16h p1 = frag_ld(Ps + wave * 2048 + mi * 16 * 64 + 32, 64);
#pragma unroll
      for (int nj = 0; nj < 4; ++nj) {
        o[mi][nj] = WMMA_F16(p0, frag_ld(VsT + nj * 16 * 64 + 0, 64), o[mi][nj]);
        o[mi][nj] = WMMA_F16(p1, frag_ld(VsT + nj * 16 * 64 + 32, 64), o[mi][nj]);
      }
    }
  }

  // O /= l ; write fp32 merged-head layout
#pragma unroll
  for (int mi = 0; mi < 2; ++mi)
#pragma unroll
    for (int nj = 0; nj < 4; ++nj)
#pragma unroll
      for (int e = 0; e < 8; ++e) {
        size_t row = qrb + wave * 32 + mi * 16 + e + 8 * hg;
        int col = h * 64 + nj * 16 + nl;
        O[row * 512 + col] = o[mi][nj][e] / lrow[mi][e];
      }
}

// ---------------------------------------------------------------------------
// LayerNorm over 512 cols; one wave32 per row, 8 rows per 256-thread block.
// ---------------------------------------------------------------------------
__global__ __launch_bounds__(256) void layernorm_kernel(
    const float* __restrict__ X, const float* __restrict__ gamma,
    const float* __restrict__ beta, float* __restrict__ Yout) {
  const int t = threadIdx.x, wave = t >> 5, lane = t & 31;
  const size_t row = (size_t)blockIdx.x * 8 + wave;
  const float* xr = X + row * 512;
  float x[16], s = 0.f, s2 = 0.f;
#pragma unroll
  for (int j = 0; j < 16; ++j) {
    x[j] = xr[j * 32 + lane];
    s += x[j];
    s2 += x[j] * x[j];
  }
#pragma unroll
  for (int off = 1; off < 32; off <<= 1) {
    s += __shfl_xor(s, off, 32);
    s2 += __shfl_xor(s2, off, 32);
  }
  float mu = s * (1.f / 512.f);
  float var = s2 * (1.f / 512.f) - mu * mu;
  float inv = rsqrtf(var + 1e-5f);
#pragma unroll
  for (int j = 0; j < 16; ++j) {
    int col = j * 32 + lane;
    Yout[row * 512 + col] = (x[j] - mu) * inv * gamma[col] + beta[col];
  }
}

// ---------------------------------------------------------------------------
extern "C" void kernel_launch(void* const* d_in, const int* in_sizes, int n_in,
                              void* d_out, int out_size, void* d_ws,
                              size_t ws_size, hipStream_t stream) {
  (void)in_sizes; (void)n_in; (void)out_size; (void)ws_size;
  const float* X    = (const float*)d_in[0];  // [4,2048,512]
  const float* Y    = (const float*)d_in[1];  // [4,2048,512]
  const float* Wq   = (const float*)d_in[2];  // [512,512]
  const float* bq   = (const float*)d_in[3];
  const float* Wkv  = (const float*)d_in[4];  // [512,1024]
  const float* bkv  = (const float*)d_in[5];
  const float* Wo   = (const float*)d_in[6];  // [512,512]
  const float* bo   = (const float*)d_in[7];
  const float* gam  = (const float*)d_in[8];
  const float* bet  = (const float*)d_in[9];

  char* ws = (char*)d_ws;
  _Float16* Qh  = (_Float16*)(ws);                          //  8 MiB f16 Q
  _Float16* KVh = (_Float16*)(ws + (size_t)(8u  << 20));    // 16 MiB f16 KV
  float*    Oat = (float*)   (ws + (size_t)(24u << 20));    // 16 MiB f32 attn out
  float*    T   = (float*)   (ws + (size_t)(40u << 20));    // 16 MiB f32 pre-LN

  const int M = 4 * 2048;  // 8192 rows
  dim3 blk(256);

  // Q = X*Wq + bq  (f16 out)
  gemm_bias_kernel<true><<<dim3(512 / 64, M / 128), blk, 0, stream>>>(
      X, 512, Wq, 512, bq, nullptr, (void*)Qh, 512, 512);
  // KV = Y*Wkv + bkv  (f16 out)
  gemm_bias_kernel<true><<<dim3(1024 / 64, M / 128), blk, 0, stream>>>(
      Y, 512, Wkv, 1024, bkv, nullptr, (void*)KVh, 1024, 512);
  // Flash attention
  attention_kernel<<<dim3(2048 / 128, 8, 4), dim3(128), 0, stream>>>(Qh, KVh, Oat);
  // T = O*Wo + bo + X  (f32 out, fused residual)
  gemm_bias_kernel<false><<<dim3(512 / 64, M / 128), blk, 0, stream>>>(
      Oat, 512, Wo, 512, bo, X, (void*)T, 512, 512);
  // LayerNorm -> d_out
  layernorm_kernel<<<dim3(M / 8), blk, 0, stream>>>(T, gam, bet, (float*)d_out);
}
